// SemanticEvolutionaryVQ_67619965108646
// MI455X (gfx1250) — matrix operand order repaired and checked by reading, not compiled, not run on hardware
//
#include <hip/hip_runtime.h>

typedef __attribute__((ext_vector_type(16))) _Float16     v16h;
typedef __attribute__((ext_vector_type(8)))  float        v8f;
typedef __attribute__((ext_vector_type(4)))  unsigned int v4u;
typedef __attribute__((ext_vector_type(8)))  unsigned int v8u;

#define HALF_DIM 64
#define DIM      128
#define N_ROWS   (32 * 2048)          // 65536 vectors
#define K_SHAPE  1024
#define K_COLOR  16

// d_out layout (all float32): quantized | vq_loss | rep_loss | s_idx | c_idx
#define Q_OFF    0
#define VQ_OFF   (N_ROWS * DIM)       // 8388608
#define REP_OFF  (VQ_OFF + 1)
#define SIDX_OFF (VQ_OFF + 2)
#define CIDX_OFF (SIDX_OFF + N_ROWS)

// LDS staging layout: padded codebook rows (TDM pad: 128B data + 32B pad)
#define ROWB     160                          // padded LDS row stride (bytes)
#define SH_OFF   0                            // shape codebook, 1024 rows
#define CL_OFF   (K_SHAPE * ROWB)             // 163840: color codebook, 16 rows
#define WSQ_OFF  (CL_OFF + K_COLOR * ROWB)    // 166400: wsq floats
#define SMEM_BYTES (WSQ_OFF + (K_SHAPE + K_COLOR) * 4)   // 170560 < 320KB/WGP

// ---------------------------------------------------------------------------
// init: zero the atomic accumulators (harness poisons ws with 0xAA)
// ---------------------------------------------------------------------------
__global__ void vq_init(float* accum) {
    if (threadIdx.x < 8) accum[threadIdx.x] = 0.0f;
}

// ---------------------------------------------------------------------------
// prep: per-code squared norms + f16 copy of codebooks (so B-fragments are
// single contiguous 32B v16h loads in the same contiguous k order as A)
// ---------------------------------------------------------------------------
__global__ void vq_prep(const float* __restrict__ Ws, const float* __restrict__ Wc,
                        float* __restrict__ wsq_s, float* __restrict__ wsq_c,
                        _Float16* __restrict__ Wh_s, _Float16* __restrict__ Wh_c) {
    int r = blockIdx.x * blockDim.x + threadIdx.x;
    if (r < K_SHAPE) {
        const float* w = Ws + r * HALF_DIM;
        _Float16* o = Wh_s + r * HALF_DIM;
        float s = 0.0f;
        for (int d = 0; d < HALF_DIM; ++d) { float v = w[d]; s += v * v; o[d] = (_Float16)v; }
        wsq_s[r] = s;
    } else if (r < K_SHAPE + K_COLOR) {
        int rr = r - K_SHAPE;
        const float* w = Wc + rr * HALF_DIM;
        _Float16* o = Wh_c + rr * HALF_DIM;
        float s = 0.0f;
        for (int d = 0; d < HALF_DIM; ++d) { float v = w[d]; s += v * v; o[d] = (_Float16)v; }
        wsq_c[rr] = s;
    }
}

// ---------------------------------------------------------------------------
// TDM: 2D f16 tensor (rows x 64) -> LDS, padded 32B per 128B row so that
// lane-strided B-fragment reads spread over 8 LDS banks instead of 2.
// Descriptor per CDNA5 ISA ch.8; inline asm is portable across toolchains.
// ---------------------------------------------------------------------------
__device__ __forceinline__ void tdm_load_codebook(unsigned lds_byte_addr,
                                                  const _Float16* gsrc,
                                                  unsigned rows) {
    unsigned long long ga = (unsigned long long)(uintptr_t)gsrc;
    v4u g0;
    g0[0] = 1u;                                              // count=1, user mode
    g0[1] = lds_byte_addr;                                   // lds_addr
    g0[2] = (unsigned)ga;                                    // global_addr[31:0]
    g0[3] = ((unsigned)(ga >> 32) & 0x01FFFFFFu)             // global_addr[56:32]
          | 0x80000000u;                                     // type=2 (bits 127:126)
    v8u g1;
    g1[0] = (1u << 16)        // data_size = 1 -> 2 bytes (f16)
          | (1u << 20)        // pad_enable
          | (4u << 22)        // pad_interval: 2^(4+1)=32 DWORDs = 128B row
          | (7u << 25);       // pad_amount: 7+1 = 8 DWORDs = 32B
    g1[1] = (HALF_DIM & 0xFFFFu) << 16;                      // tensor_dim0 = 64
    g1[2] = (rows & 0xFFFFu) << 16;                          // tensor_dim1
    g1[3] = (HALF_DIM & 0xFFFFu) << 16;                      // tile_dim0 = 64
    g1[4] = rows & 0xFFFFu;                                  // tile_dim1, tile_dim2=0
    g1[5] = HALF_DIM;                                        // tensor_dim0_stride
    g1[6] = 0u;
    g1[7] = 0u;
    asm volatile("tensor_load_to_lds %0, %1" :: "s"(g0), "s"(g1) : "memory");
}

// A-fragment loader: lane (m = lane&15, hi = lane>>4) holds K = hi*16 + {0..15}
// (contiguous order, matching the B fragments).
__device__ __forceinline__ void load_a_frag(const float* __restrict__ xr, int base, int hi, v16h& a) {
    const float* p = xr + base + hi * 16;
    float4 f0 = *(const float4*)(p + 0);
    float4 f1 = *(const float4*)(p + 4);
    float4 f2 = *(const float4*)(p + 8);
    float4 f3 = *(const float4*)(p + 12);
    a[0]  = (_Float16)f0.x; a[1]  = (_Float16)f0.y; a[2]  = (_Float16)f0.z; a[3]  = (_Float16)f0.w;
    a[4]  = (_Float16)f1.x; a[5]  = (_Float16)f1.y; a[6]  = (_Float16)f1.z; a[7]  = (_Float16)f1.w;
    a[8]  = (_Float16)f2.x; a[9]  = (_Float16)f2.y; a[10] = (_Float16)f2.z; a[11] = (_Float16)f2.w;
    a[12] = (_Float16)f3.x; a[13] = (_Float16)f3.y; a[14] = (_Float16)f3.z; a[15] = (_Float16)f3.w;
}

// ---------------------------------------------------------------------------
// main: TDM-stage both codebooks into LDS once per block, then fused
// dual-codebook nearest-neighbor + gather + MSE partials.
// 8 waves/block, 16 rows/wave, 128 rows/block, 512 blocks.
// ---------------------------------------------------------------------------
__global__ __launch_bounds__(256)
void vq_main(const float* __restrict__ x,
             const float* __restrict__ Ws, const float* __restrict__ Wc,
             const float* __restrict__ wsq_s, const float* __restrict__ wsq_c,
             const _Float16* __restrict__ Wh_s, const _Float16* __restrict__ Wh_c,
             float* __restrict__ out, float* __restrict__ accum) {
    extern __shared__ char smem[];
    float* swsq = (float*)(smem + WSQ_OFF);

    const int wave = threadIdx.x >> 5;
    const int lane = threadIdx.x & 31;
    const int mrow = lane & 15;
    const int hi   = lane >> 4;
    const int tileBase = (blockIdx.x * 8 + wave) * 16;

    // --- stage codebooks (wave 0 drives the TDM) + wsq (cooperative) ---
    if (wave == 0) {
        unsigned base = (unsigned)(uintptr_t)smem;
        tdm_load_codebook(base + SH_OFF, Wh_s, K_SHAPE);
        tdm_load_codebook(base + CL_OFF, Wh_c, K_COLOR);
        __builtin_amdgcn_s_wait_tensorcnt(0);
    }
    for (int i = threadIdx.x; i < K_SHAPE + K_COLOR; i += 256)
        swsq[i] = (i < K_SHAPE) ? wsq_s[i] : wsq_c[i - K_SHAPE];
    __syncthreads();

    const float* xr = x + (size_t)(tileBase + mrow) * DIM;

    // A fragments: shape half (K 0..63) and color half (K 64..127)
    v16h a0, a1, ac0, ac1;
    load_a_frag(xr,  0, hi, a0);
    load_a_frag(xr, 32, hi, a1);
    load_a_frag(xr, 64, hi, ac0);
    load_a_frag(xr, 96, hi, ac1);

    float minv[8]; int mini[8];
#pragma unroll
    for (int r = 0; r < 8; ++r) { minv[r] = 3.4e38f; mini[r] = 0; }

    // ---- shape codebook: 64 column tiles of 16 codes, K=64 in two WMMA ----
    for (int nt = 0; nt < K_SHAPE / 16; ++nt) {
        const int ncol = nt * 16 + mrow;
        const char* bp = smem + SH_OFF + ncol * ROWB + hi * 32;   // 32B aligned
        v16h b0 = *(const v16h*)(bp);        // K = hi*16 + 0..15
        v16h b1 = *(const v16h*)(bp + 64);   // K = 32 + hi*16 + 0..15
        v8f c = {};
        c = __builtin_amdgcn_wmma_f32_16x16x32_f16(false, a0, false, b0, (short)0, c, false, false);
        c = __builtin_amdgcn_wmma_f32_16x16x32_f16(false, a1, false, b1, (short)0, c, false, false);
        const float wq = swsq[ncol];
#pragma unroll
        for (int r = 0; r < 8; ++r) {
            float d = wq - 2.0f * c[r];      // ||x||^2 constant per row: drop it
            if (d < minv[r]) { minv[r] = d; mini[r] = ncol; }
        }
    }
#pragma unroll
    for (int r = 0; r < 8; ++r) {
#pragma unroll
        for (int off = 8; off >= 1; off >>= 1) {
            float ov = __shfl_xor(minv[r], off, 32);
            int   oi = __shfl_xor(mini[r], off, 32);
            if (ov < minv[r]) { minv[r] = ov; mini[r] = oi; }
        }
    }

    // ---- color codebook: single 16-wide tile ----
    float cminv[8]; int cmini[8];
    {
        const int ncol = mrow;
        const char* bp = smem + CL_OFF + ncol * ROWB + hi * 32;
        v16h b0 = *(const v16h*)(bp);
        v16h b1 = *(const v16h*)(bp + 64);
        v8f c = {};
        c = __builtin_amdgcn_wmma_f32_16x16x32_f16(false, ac0, false, b0, (short)0, c, false, false);
        c = __builtin_amdgcn_wmma_f32_16x16x32_f16(false, ac1, false, b1, (short)0, c, false, false);
        const float wq = swsq[K_SHAPE + ncol];
#pragma unroll
        for (int r = 0; r < 8; ++r) { cminv[r] = wq - 2.0f * c[r]; cmini[r] = ncol; }
    }
#pragma unroll
    for (int r = 0; r < 8; ++r) {
#pragma unroll
        for (int off = 8; off >= 1; off >>= 1) {
            float ov = __shfl_xor(cminv[r], off, 32);
            int   oi = __shfl_xor(cmini[r], off, 32);
            if (ov < cminv[r]) { cminv[r] = ov; cmini[r] = oi; }
        }
    }

    // ---- writers: lane0 owns C-rows 0..7, lane16 owns rows 8..15 ----
    if (mrow == 0) {
        float lossAcc = 0.0f;
#pragma unroll 1
        for (int r = 0; r < 8; ++r) {
            const int m  = tileBase + hi * 8 + r;
            const int si = mini[r];
            const int ci = cmini[r];
            out[SIDX_OFF + m] = (float)si;
            out[CIDX_OFF + m] = (float)ci;
            const float* wsrow = Ws + si * HALF_DIM;
            const float* wcrow = Wc + ci * HALF_DIM;
            const float* xrow  = x   + (size_t)m * DIM;
            float*       qrow  = out + (size_t)m * DIM;
            for (int d = 0; d < HALF_DIM; ++d) {
                float qs = wsrow[d]; qrow[d] = qs;
                float ds = qs - xrow[d];            lossAcc += ds * ds;
                float qc = wcrow[d]; qrow[HALF_DIM + d] = qc;
                float dc = qc - xrow[HALF_DIM + d]; lossAcc += dc * dc;
            }
        }
        atomicAdd(&accum[0], lossAcc);
    }
}

// ---------------------------------------------------------------------------
// repulsion: one block per code row i; sum 1/(d_ij + 1e-4) incl. diagonal
// ---------------------------------------------------------------------------
__global__ __launch_bounds__(256)
void vq_rep(const float* __restrict__ W, const float* __restrict__ wsq,
            int K, float* accum_slot) {
    __shared__ float red[256];
    const int i = blockIdx.x;
    const float* wi = W + i * HALF_DIM;
    const float sqi = wsq[i];
    float s = 0.0f;
    for (int j = threadIdx.x; j < K; j += blockDim.x) {
        const float* wj = W + j * HALF_DIM;
        float dot = 0.0f;
        for (int d = 0; d < HALF_DIM; ++d) dot += wi[d] * wj[d];
        float dist = sqi + wsq[j] - 2.0f * dot;
        if (j == i) dist += 1e-5f;
        s += 1.0f / (dist + 1e-4f);
    }
    red[threadIdx.x] = s;
    __syncthreads();
    for (int st = 128; st > 0; st >>= 1) {
        if ((int)threadIdx.x < st) red[threadIdx.x] += red[threadIdx.x + st];
        __syncthreads();
    }
    if (threadIdx.x == 0) atomicAdd(accum_slot, red[0]);
}

// ---------------------------------------------------------------------------
// finalize scalars
// ---------------------------------------------------------------------------
__global__ void vq_fin(const float* __restrict__ accum, float* __restrict__ out) {
    float mse = accum[0] / (float)(N_ROWS * DIM);
    out[VQ_OFF] = 1.25f * mse;   // q_loss + 0.25*e_loss, equal numerically
    float repS = (accum[1] - (float)K_SHAPE * 1e4f) / ((float)K_SHAPE * (float)(K_SHAPE - 1));
    float repC = (accum[2] - (float)K_COLOR * 1e4f) / ((float)K_COLOR * (float)(K_COLOR - 1));
    out[REP_OFF] = 0.1f * (repS + repC);
}

// ---------------------------------------------------------------------------
extern "C" void kernel_launch(void* const* d_in, const int* in_sizes, int n_in,
                              void* d_out, int out_size, void* d_ws, size_t ws_size,
                              hipStream_t stream) {
    const float* x  = (const float*)d_in[0];   // [32,2048,128]
    const float* Ws = (const float*)d_in[1];   // [1024,64]
    const float* Wc = (const float*)d_in[2];   // [16,64]
    float* out = (float*)d_out;

    // workspace: wsq_s[1024] | wsq_c[16] | accum[8] | Wh_s f16[1024*64] | Wh_c f16[16*64]
    float* wsq_s = (float*)d_ws;
    float* wsq_c = wsq_s + K_SHAPE;
    float* accum = wsq_c + K_COLOR;
    _Float16* Wh_s = (_Float16*)(accum + 8);             // byte off 4192, 32B aligned
    _Float16* Wh_c = Wh_s + (size_t)K_SHAPE * HALF_DIM;

    vq_init<<<1, 32, 0, stream>>>(accum);
    vq_prep<<<(K_SHAPE + K_COLOR + 255) / 256, 256, 0, stream>>>(Ws, Wc, wsq_s, wsq_c, Wh_s, Wh_c);
    vq_main<<<N_ROWS / 128, 256, SMEM_BYTES, stream>>>(x, Ws, Wc, wsq_s, wsq_c, Wh_s, Wh_c, out, accum);
    vq_rep<<<K_SHAPE, 256, 0, stream>>>(Ws, wsq_s, K_SHAPE, accum + 1);
    vq_rep<<<K_COLOR, 256, 0, stream>>>(Wc, wsq_c, K_COLOR, accum + 2);
    vq_fin<<<1, 1, 0, stream>>>(accum, out);
}
